// FusedMultiHeadAttention_90941637525735
// MI455X (gfx1250) — matrix operand order repaired and checked by reading, not compiled, not run on hardware
//
#include <hip/hip_runtime.h>
#include <hip/hip_bf16.h>

// ---------------------------------------------------------------------------
// Fused multi-head attention for MI455X (gfx1250), bf16 WMMA pipeline with
// double-buffered TDM (tensor_load_to_lds) staging and permlane16 softmax.
//   B=4, L=2048, D=1024, H=16, Hd=64
// ---------------------------------------------------------------------------

#define BATCH   4
#define SEQ     2048
#define EMBED   1024
#define HEADS   16
#define HDIM    64
#define QKVDIM  (3 * EMBED)
#define MROWS   (BATCH * SEQ)          // 8192
#define ATTN_SCALE 0.125f              // 1/sqrt(64)

typedef __attribute__((ext_vector_type(16))) __bf16 v16bf;
typedef __attribute__((ext_vector_type(8)))  float  v8f;
typedef unsigned int v4u __attribute__((ext_vector_type(4)));
typedef int          v8s __attribute__((ext_vector_type(8)));
typedef int          v4s __attribute__((ext_vector_type(4)));

union ABf {                 // 16 bf16 A/B fragment <-> 8 dwords
    v16bf v;
    unsigned int u[8];
};
union CFr {                 // 8 f32 C/D fragment
    v8f v;
    float f[8];
};

static __device__ __forceinline__ unsigned short f32_to_bf16(float x) {
    unsigned int u = __builtin_bit_cast(unsigned int, x);
    u += 0x7FFFu + ((u >> 16) & 1u);   // round-to-nearest-even
    return (unsigned short)(u >> 16);
}

// ---- VALU 16-lane butterfly reductions via v_permlane16_b32 ---------------
static __device__ __forceinline__ float permf16(float v, unsigned s0, unsigned s1) {
    unsigned u = __builtin_bit_cast(unsigned, v);
    unsigned r = __builtin_amdgcn_permlane16(u, u, s0, s1, false, false);
    return __builtin_bit_cast(float, r);
}
static __device__ __forceinline__ float red_max16(float v) {
    v = fmaxf(v, permf16(v, 0x67452301u, 0xEFCDAB89u));   // xor 1
    v = fmaxf(v, permf16(v, 0x54761032u, 0xDCFE98BAu));   // xor 2
    v = fmaxf(v, permf16(v, 0x32107654u, 0xBA98FEDCu));   // xor 4
    v = fmaxf(v, permf16(v, 0xFEDCBA98u, 0x76543210u));   // xor 8
    return v;
}
static __device__ __forceinline__ float red_sum16(float v) {
    v += permf16(v, 0x67452301u, 0xEFCDAB89u);
    v += permf16(v, 0x54761032u, 0xDCFE98BAu);
    v += permf16(v, 0x32107654u, 0xBA98FEDCu);
    v += permf16(v, 0xFEDCBA98u, 0x76543210u);
    return v;
}

// ---- Tensor Data Mover: 2D bf16 tile, global -> LDS -----------------------
// D# per CDNA5 ISA ch.8: group0 {count=1, lds_addr, global_addr, type=2},
// group1 {data_size=2B, tensor_dim0/1, tile_dim0/1, tensor_dim0_stride}.
// 6-arg builtin form (clang-23 / therock headers).
static __device__ __forceinline__ void tdm_load_tile_2d_bf16(
    const unsigned short* gptr,    // global tile start
    unsigned lds_off,              // LDS byte offset of destination
    unsigned tensor_d0,            // tensor row length (elems)
    unsigned tensor_d1,            // tensor #rows
    unsigned tile_d0,              // tile width (elems)
    unsigned tile_d1,              // tile height (rows)
    unsigned stride0)              // row stride (elems)
{
    unsigned long long ga = (unsigned long long)gptr;
    v4u g0;
    g0[0] = 1u;                                            // count=1
    g0[1] = lds_off;                                       // lds_addr
    g0[2] = (unsigned)(ga & 0xffffffffu);                  // global_addr lo
    g0[3] = (unsigned)((ga >> 32) & 0x01ffffffu) | (2u << 30); // hi + type=2
    v8s g1;
    g1[0] = (int)(1u << 16);                               // data_size = 2B
    g1[1] = (int)((tensor_d0 & 0xffffu) << 16);            // td0[15:0]
    g1[2] = (int)(((tensor_d0 >> 16) & 0xffffu) |
                  ((tensor_d1 & 0xffffu) << 16));          // td0 hi | td1 lo
    g1[3] = (int)(((tensor_d1 >> 16) & 0xffffu) |
                  ((tile_d0 & 0xffffu) << 16));            // td1 hi | tile0
    g1[4] = (int)(tile_d1 & 0xffffu);                      // tile1 (tile2=0)
    g1[5] = (int)stride0;                                  // stride0 lo32
    g1[6] = 0;                                             // stride0 hi / stride1 lo
    g1[7] = 0;
    v4s z4 = {0, 0, 0, 0};
    v8s z8 = {0, 0, 0, 0, 0, 0, 0, 0};
    __builtin_amdgcn_tensor_load_to_lds(g0, g1, z4, z4, z8, 0);
}

// ---------------------------------------------------------------------------
// Kernel 1: packed fp32 -> bf16 conversion (4 elems / thread, exact grids)
// ---------------------------------------------------------------------------
__global__ __launch_bounds__(256)
void cvt_f32_to_bf16(const float* __restrict__ src,
                     unsigned short* __restrict__ dst) {
    const size_t i = ((size_t)blockIdx.x * blockDim.x + threadIdx.x) * 4;
    float4 f = *(const float4*)(src + i);
    unsigned int lo = (unsigned int)f32_to_bf16(f.x) |
                      ((unsigned int)f32_to_bf16(f.y) << 16);
    unsigned int hi = (unsigned int)f32_to_bf16(f.z) |
                      ((unsigned int)f32_to_bf16(f.w) << 16);
    uint2 p; p.x = lo; p.y = hi;
    *(uint2*)(dst + i) = p;
}

// ---------------------------------------------------------------------------
// Kernel 2/4: tiled bf16 GEMM,  C[M,N] = A[M,K] * B[K,N] + bias
//   block: 256 threads (8 waves), tile 128(M) x 64(N), K-step 32
//   Double-buffered: A tile via TDM, B tile transposed through registers.
//   One barrier per K-step; DMA/global latency hidden behind WMMA.
//   MODE 0: scatter bf16 into Q/K/V [B,H,L,Hd]   MODE 1: plain f32 store
// ---------------------------------------------------------------------------
template <int MODE>
__global__ __launch_bounds__(256)
void gemm_bf16_wmma(const unsigned short* __restrict__ A,
                    const unsigned short* __restrict__ Bm,
                    const float* __restrict__ bias,
                    void* __restrict__ out,
                    int Ncols, int Kdim) {
    __shared__ unsigned short As[2][128 * 32];   // [m][k] row-major, ping-pong
    __shared__ unsigned short Bs[2][64 * 32];    // [n][k] transposed, ping-pong

    const int t    = threadIdx.x;
    const int wave = t >> 5;
    const int lane = t & 31;
    const int hl   = lane >> 4;                // K-half selector (wave32 WMMA layout)
    const int l16  = lane & 15;
    const int wm   = wave >> 1;                // 4 waves along M
    const int wn   = wave & 1;                 // 2 waves along N
    const int blockM = blockIdx.y * 128;
    const int blockN = blockIdx.x * 64;
    const int kIters = Kdim >> 5;
    const int bk = t >> 3;                     // staging: k row 0..31
    const int bn = (t & 7) * 8;                // staging: n group of 8

    // ---- prologue: stage tile 0
    {
        if (wave == 0)
            tdm_load_tile_2d_bf16(A + (size_t)blockM * Kdim,
                                  (unsigned)(size_t)As[0],
                                  (unsigned)Kdim, MROWS, 32, 128, (unsigned)Kdim);
        uint4 val = *(const uint4*)(Bm + (size_t)bk * Ncols + blockN + bn);
        const unsigned short* pv = (const unsigned short*)&val;
        #pragma unroll
        for (int i = 0; i < 8; ++i)
            Bs[0][(bn + i) * 32 + bk] = pv[i];
        if (wave == 0) __builtin_amdgcn_s_wait_tensorcnt(0);
    }
    __syncthreads();

    v8f acc[2][2] = {};

    for (int it = 0; it < kIters; ++it) {
        const int cur = it & 1;
        const unsigned short* Ac = As[cur];
        const unsigned short* Bc = Bs[cur];
        const bool do_stage = (it + 1 < kIters);

        // ---- 1) fragment loads from current buffers (issued first: LDS is
        //         in-order per wave, keep them ahead of staging stores)
        ABf af[2];
        #pragma unroll
        for (int i = 0; i < 2; ++i) {
            const unsigned int* row =
                (const unsigned int*)(Ac + (wm * 32 + i * 16 + l16) * 32);
            #pragma unroll
            for (int j = 0; j < 8; ++j) {
                const int kk = ((j >> 2) << 4) + hl * 8 + ((j & 3) << 1);
                af[i].u[j] = row[kk >> 1];
            }
        }
        ABf bfr[2];
        #pragma unroll
        for (int j = 0; j < 2; ++j) {
            const unsigned int* col =
                (const unsigned int*)(Bc + (wn * 32 + j * 16 + l16) * 32);
            #pragma unroll
            for (int e = 0; e < 8; ++e)
                bfr[j].u[e] = col[(((e >> 2) << 4) + hl * 8 + ((e & 3) << 1)) >> 1];
        }

        // ---- 2) kick off next tile: TDM for A, global load for B
        uint4 bval = {};
        if (do_stage) {
            const int k0n = (it + 1) << 5;
            if (wave == 0)
                tdm_load_tile_2d_bf16(A + (size_t)blockM * Kdim + k0n,
                                      (unsigned)(size_t)As[cur ^ 1],
                                      (unsigned)Kdim, MROWS, 32, 128,
                                      (unsigned)Kdim);
            bval = *(const uint4*)(Bm + (size_t)(k0n + bk) * Ncols + blockN + bn);
            if (it + 2 < kIters)   // prefetch B tile after next into GL2
                __builtin_prefetch(
                    Bm + (size_t)(((it + 2) << 5) + bk) * Ncols + blockN + bn, 0, 3);
        }

        // ---- 3) WMMA on current tile (overlaps DMA / global latency)
        #pragma unroll
        for (int i = 0; i < 2; ++i)
            #pragma unroll
            for (int j = 0; j < 2; ++j)
                acc[i][j] = __builtin_amdgcn_wmma_f32_16x16x32_bf16(
                    false, af[i].v, false, bfr[j].v, (short)0, acc[i][j],
                    false, false);

        // ---- 4) finish staging: transpose-store B into the other buffer
        if (do_stage) {
            const unsigned short* pv = (const unsigned short*)&bval;
            #pragma unroll
            for (int i = 0; i < 8; ++i)
                Bs[cur ^ 1][(bn + i) * 32 + bk] = pv[i];
            if (wave == 0) __builtin_amdgcn_s_wait_tensorcnt(0);
        }
        __syncthreads();
    }

    // ---- epilogue
    #pragma unroll
    for (int i = 0; i < 2; ++i) {
        #pragma unroll
        for (int j = 0; j < 2; ++j) {
            CFr c; c.v = acc[i][j];
            const int n  = blockN + wn * 32 + j * 16 + l16;
            const float bv = bias[n];
            #pragma unroll
            for (int r = 0; r < 8; ++r) {
                const int m = blockM + wm * 32 + i * 16 + hl * 8 + r;
                const float val = c.f[r] + bv;
                if (MODE == 0) {
                    // scatter into Q/K/V [B,H,L,Hd] bf16
                    const int which = n >> 10;          // 0=Q 1=K 2=V
                    const int dcol  = n & 1023;
                    const int h  = dcol >> 6;
                    const int hd = dcol & 63;
                    const int b  = m >> 11;
                    const int l  = m & 2047;
                    unsigned short* q = (unsigned short*)out;
                    const size_t base = (size_t)which * ((size_t)BATCH * HEADS * SEQ * HDIM);
                    q[base + ((((size_t)b * HEADS + h) * SEQ + l) * HDIM + hd)] =
                        f32_to_bf16(val);
                } else {
                    ((float*)out)[(size_t)m * Ncols + n] = val;
                }
            }
        }
    }
}

// ---------------------------------------------------------------------------
// Kernel 3: flash attention per (b,h), double-buffered key/value chunks.
//   grid: x = L/128 query blocks, y = B*H
//   block: 256 threads = 8 waves; each wave owns 16 query rows, full Hd=64.
//   K chunk via TDM, V chunk transposed through registers; one barrier/chunk.
// ---------------------------------------------------------------------------
__global__ __launch_bounds__(256)
void flash_attn_wmma(const unsigned short* __restrict__ Q,
                     const unsigned short* __restrict__ Kb,
                     const unsigned short* __restrict__ V,
                     unsigned short* __restrict__ Obf) {
    __shared__ unsigned short Kc[2][32 * 64];      // [key][d], ping-pong
    __shared__ unsigned short Vt[2][64 * 32];      // [d][key], ping-pong
    __shared__ unsigned short Ps[8][16 * 32];      // per-wave P scratch [q][key]

    const int t    = threadIdx.x;
    const int wave = t >> 5;
    const int lane = t & 31;
    const int hl   = lane >> 4;
    const int l16  = lane & 15;
    const int bh   = blockIdx.y;
    const size_t head_base = (size_t)bh * SEQ * HDIM;
    const int q0 = blockIdx.x * 128 + wave * 16;
    const int sk = t >> 3;                         // staging: key row 0..31
    const int sd = (t & 7) * 8;                    // staging: d group of 8

    // ---- Q fragments for this wave's 16 rows (2 K-chunks over Hd=64)
    ABf qf[2];
    {
        const unsigned int* row =
            (const unsigned int*)(Q + head_base + (size_t)(q0 + l16) * HDIM);
        #pragma unroll
        for (int dc = 0; dc < 2; ++dc)
            #pragma unroll
            for (int j = 0; j < 8; ++j) {
                const int d = dc * 32 + ((j >> 2) << 4) + hl * 8 + ((j & 3) << 1);
                qf[dc].u[j] = row[d >> 1];
            }
    }

    // ---- prologue: stage chunk 0
    {
        if (wave == 0)
            tdm_load_tile_2d_bf16(Kb + head_base, (unsigned)(size_t)Kc[0],
                                  HDIM, SEQ, HDIM, 32, HDIM);
        uint4 val = *(const uint4*)(V + head_base + (size_t)sk * HDIM + sd);
        const unsigned short* pv = (const unsigned short*)&val;
        #pragma unroll
        for (int i = 0; i < 8; ++i)
            Vt[0][(sd + i) * 32 + sk] = pv[i];
        if (wave == 0) __builtin_amdgcn_s_wait_tensorcnt(0);
    }
    __syncthreads();

    v8f oacc[4] = {};                 // 16q x 64d output accumulators
    float rm[8], rl[8];
    #pragma unroll
    for (int r = 0; r < 8; ++r) { rm[r] = -1e30f; rl[r] = 0.0f; }

    const int nIter = SEQ / 32;
    for (int it = 0; it < nIter; ++it) {
        const int cur = it & 1;
        const bool do_stage = (it + 1 < nIter);

        // ---- 1) K^T fragments from current K chunk
        ABf kf[2][2];
        #pragma unroll
        for (int kk = 0; kk < 2; ++kk) {
            const unsigned int* col =
                (const unsigned int*)(Kc[cur] + (kk * 16 + l16) * 64);
            #pragma unroll
            for (int dc = 0; dc < 2; ++dc)
                #pragma unroll
                for (int e = 0; e < 8; ++e)
                    kf[kk][dc].u[e] = col[(dc * 32 + ((e >> 2) << 4) + hl * 8 + ((e & 3) << 1)) >> 1];
        }
        // ---- 2) V fragments from current V chunk (before staging stores)
        ABf vf[4];
        #pragma unroll
        for (int j = 0; j < 4; ++j) {
            const unsigned int* col =
                (const unsigned int*)(Vt[cur] + (j * 16 + l16) * 32);
            #pragma unroll
            for (int e = 0; e < 8; ++e)
                vf[j].u[e] = col[(((e >> 2) << 4) + hl * 8 + ((e & 3) << 1)) >> 1];
        }

        // ---- 3) kick off next chunk: TDM for K, global load for V
        uint4 sval = {};
        if (do_stage) {
            const int kbn = (it + 1) * 32;
            if (wave == 0)
                tdm_load_tile_2d_bf16(Kb + head_base + (size_t)kbn * HDIM,
                                      (unsigned)(size_t)Kc[cur ^ 1],
                                      HDIM, SEQ, HDIM, 32, HDIM);
            sval = *(const uint4*)(V + head_base + (size_t)(kbn + sk) * HDIM + sd);
        }

        // ---- 4) scores S = Q * K^T  (16q x 32keys)
        v8f sacc[2] = {};
        #pragma unroll
        for (int kk = 0; kk < 2; ++kk)
            #pragma unroll
            for (int dc = 0; dc < 2; ++dc)
                sacc[kk] = __builtin_amdgcn_wmma_f32_16x16x32_bf16(
                    false, qf[dc].v, false, kf[kk][dc].v, (short)0, sacc[kk],
                    false, false);

        // ---- 5) streaming softmax (row-wise over 16 lanes, pure VALU)
        CFr c0, c1; c0.v = sacc[0]; c1.v = sacc[1];
        float alpha[8];
        #pragma unroll
        for (int r = 0; r < 8; ++r) {
            const float s0 = c0.f[r] * ATTN_SCALE;
            const float s1 = c1.f[r] * ATTN_SCALE;
            const float mx = red_max16(fmaxf(s0, s1));
            const float nm = fmaxf(rm[r], mx);
            alpha[r] = __expf(rm[r] - nm);
            rm[r] = nm;
            const float p0 = __expf(s0 - nm);
            const float p1 = __expf(s1 - nm);
            c0.f[r] = p0; c1.f[r] = p1;
            rl[r] = rl[r] * alpha[r] + red_sum16(p0 + p1);
        }
        #pragma unroll
        for (int j = 0; j < 4; ++j) {
            CFr o; o.v = oacc[j];
            #pragma unroll
            for (int r = 0; r < 8; ++r) o.f[r] *= alpha[r];
            oacc[j] = o.v;
        }

        // ---- 6) re-stripe P: C-layout -> A-layout via wave-private LDS
        unsigned short* P = Ps[wave];
        #pragma unroll
        for (int r = 0; r < 8; ++r) {
            P[(hl * 8 + r) * 32 + l16]      = f32_to_bf16(c0.f[r]);
            P[(hl * 8 + r) * 32 + 16 + l16] = f32_to_bf16(c1.f[r]);
        }
        ABf pf;
        {
            const unsigned int* row = (const unsigned int*)(P + l16 * 32);
            #pragma unroll
            for (int j = 0; j < 8; ++j) {
                const int kk = ((j >> 2) << 4) + hl * 8 + ((j & 3) << 1);
                pf.u[j] = row[kk >> 1];
            }
        }

        // ---- 7) finish staging: transpose-store V into the other buffer
        if (do_stage) {
            const unsigned short* pv = (const unsigned short*)&sval;
            #pragma unroll
            for (int i = 0; i < 8; ++i)
                Vt[cur ^ 1][(sd + i) * 32 + sk] = pv[i];
        }

        // ---- 8) O += P * V  (V fragments were preloaded in step 2)
        #pragma unroll
        for (int j = 0; j < 4; ++j)
            oacc[j] = __builtin_amdgcn_wmma_f32_16x16x32_bf16(
                false, pf.v, false, vf[j].v, (short)0, oacc[j], false, false);

        // ---- 9) close the pipeline stage
        if (do_stage) {
            if (wave == 0) __builtin_amdgcn_s_wait_tensorcnt(0);
            __syncthreads();
        }
    }

    // ---- epilogue: normalize and write bf16 into [B*L, D] (head-merge fused)
    const int h = bh & (HEADS - 1);
    const int b = bh >> 4;
    #pragma unroll
    for (int j = 0; j < 4; ++j) {
        CFr o; o.v = oacc[j];
        #pragma unroll
        for (int r = 0; r < 8; ++r) {
            const int qrow = q0 + hl * 8 + r;
            const int d    = j * 16 + l16;
            Obf[((size_t)(b * SEQ + qrow)) * EMBED + h * HDIM + d] =
                f32_to_bf16(o.f[r] / rl[r]);
        }
    }
}

// ---------------------------------------------------------------------------
// Host-side launcher
// ---------------------------------------------------------------------------
extern "C" void kernel_launch(void* const* d_in, const int* in_sizes, int n_in,
                              void* d_out, int out_size, void* d_ws, size_t ws_size,
                              hipStream_t stream) {
    (void)in_sizes; (void)n_in; (void)out_size; (void)ws_size;

    const float* x     = (const float*)d_in[0];   // [4,2048,1024]
    const float* W_qkv = (const float*)d_in[1];   // [1024,3072]
    const float* b_qkv = (const float*)d_in[2];   // [3072]
    const float* W_out = (const float*)d_in[3];   // [1024,1024]
    const float* b_out = (const float*)d_in[4];   // [1024]
    float* out = (float*)d_out;                   // [4,2048,1024]

    // workspace partition (bf16 buffers, element counts)
    const size_t N_X    = (size_t)MROWS * EMBED;        // 8388608
    const size_t N_WQKV = (size_t)EMBED * QKVDIM;       // 3145728
    const size_t N_WOUT = (size_t)EMBED * EMBED;        // 1048576
    unsigned short* xb    = (unsigned short*)d_ws;
    unsigned short* wqkvb = xb + N_X;
    unsigned short* woutb = wqkvb + N_WQKV;
    unsigned short* qkv   = woutb + N_WOUT;             // Q,K,V each N_X
    unsigned short* attnb = qkv + 3 * N_X;

    // 1) fp32 -> bf16 conversions
    cvt_f32_to_bf16<<<(int)(N_X    / 1024), 256, 0, stream>>>(x,     xb);
    cvt_f32_to_bf16<<<(int)(N_WQKV / 1024), 256, 0, stream>>>(W_qkv, wqkvb);
    cvt_f32_to_bf16<<<(int)(N_WOUT / 1024), 256, 0, stream>>>(W_out, woutb);

    // 2) QKV projection + head split: [8192,1024] x [1024,3072] -> Q/K/V bf16
    gemm_bf16_wmma<0><<<dim3(QKVDIM / 64, MROWS / 128), 256, 0, stream>>>(
        xb, wqkvb, b_qkv, qkv, QKVDIM, EMBED);

    // 3) flash attention per (b,h)
    flash_attn_wmma<<<dim3(SEQ / 128, BATCH * HEADS), 256, 0, stream>>>(
        qkv, qkv + N_X, qkv + 2 * N_X, attnb);

    // 4) output projection: [8192,1024] x [1024,1024] + bias -> f32 out
    gemm_bf16_wmma<1><<<dim3(EMBED / 64, MROWS / 128), 256, 0, stream>>>(
        attnb, woutb, b_out, out, EMBED, EMBED);
}
